// Concare_24086176596192
// MI455X (gfx1250) — matrix operand order repaired and checked by reading, not compiled, not run on hardware
//
#include <hip/hip_runtime.h>

typedef __bf16 bf16_t;
typedef __bf16 bf16x8  __attribute__((ext_vector_type(8)));
typedef __bf16 bf16x16 __attribute__((ext_vector_type(16)));
typedef float  f32x8   __attribute__((ext_vector_type(8)));

#define NB 64    // batch
#define NT 64    // time
#define NI 64    // features
#define NH 256   // hidden
#define G3 768   // 3*NH
#define HSTR 264 // bf16 row stride for h in LDS (256 + 8 pad -> conflict-free b128 loads)

// fast math: hardware v_rcp_f32 / v_exp_f32 instead of IEEE div sequences
__device__ __forceinline__ float fast_rcp(float x) { return __builtin_amdgcn_rcpf(x); }
__device__ __forceinline__ float fast_sigmoid(float x) {
  return fast_rcp(1.0f + __expf(-x));            // v_exp_f32 + v_rcp_f32
}
__device__ __forceinline__ float fast_tanh(float x) {
  return 1.0f - 2.0f * fast_rcp(__expf(2.0f * x) + 1.0f);
}

// ---------------- fp32 -> bf16 weight conversion (grid-stride) ----------------
__global__ void __launch_bounds__(256)
cvt_bf16_kernel(const float* __restrict__ src, bf16_t* __restrict__ dst, int n) {
  int i = blockIdx.x * blockDim.x + threadIdx.x;
  const int stride = gridDim.x * blockDim.x;
  for (; i < n; i += stride) dst[i] = (bf16_t)src[i];
}

// ---------------- per-feature GRU with bf16 WMMA ----------------
// grid = NI blocks (one feature each), 512 threads = 16 waves.
// Wave w owns gate-column block [16w,16w+16): r-tile = ntile w, z = w+16, n = w+32,
// so r/z/n for one (b,hc) sit in the same lane -> register-local gate update.
// K loop kept rolled (#pragma unroll 1): live set = 96 acc + 32 hreg + 32 frag
// VGPRs, under the 256-VGPR cap -> no scratch spills in the hot loop.
__global__ void __launch_bounds__(512, 1)
gru_wmma_kernel(const float* __restrict__ x,     // [B,T,I]
                const int*   __restrict__ lens,  // [B]
                const float* __restrict__ wi,    // [I,3H]
                const float* __restrict__ bi,    // [I,3H]
                const bf16_t* __restrict__ whb,  // [I,3H,H] bf16
                const float* __restrict__ bh,    // [I,3H]
                float* __restrict__ ht)          // [B,I,H] out
{
  __shared__ __align__(16) bf16_t hsh[NB * HSTR];
  __shared__ float xcol[NB];
  __shared__ int   lsh[NB];

  const int i    = blockIdx.x;
  const int tid  = threadIdx.x;
  const int lane = tid & 31;
  const int wave = tid >> 5;          // 0..15
  const int lm   = lane & 15;
  const int rofs = (lane >> 4) << 3;  // 0 or 8 (C/D row offset for upper lanes)
  const int hc   = wave * 16 + lm;    // this lane's hidden column (fixed)

  if (tid < NB) lsh[tid] = lens[tid];
  for (int idx = tid; idx < NB * HSTR; idx += 512) hsh[idx] = (bf16_t)0.0f;

  // gate parameters for column hc (loop-invariant -> registers)
  const float wir = wi[i*G3 + hc],       bir  = bi[i*G3 + hc],       bhr = bh[i*G3 + hc];
  const float wiz = wi[i*G3 + hc + 256], biz  = bi[i*G3 + hc + 256], bhz = bh[i*G3 + hc + 256];
  const float win = wi[i*G3 + hc + 512], bin_ = bi[i*G3 + hc + 512], bhn = bh[i*G3 + hc + 512];

  float hreg[32];                      // fp32 master copy of owned h elements
  #pragma unroll
  for (int e = 0; e < 32; ++e) hreg[e] = 0.0f;

  // B-operand row bases: column g = ntile*16 + lm, contiguous over k
  const bf16_t* wB0 = whb + (size_t)(i*G3 + (wave      )*16 + lm) * NH;
  const bf16_t* wB1 = whb + (size_t)(i*G3 + (wave + 16 )*16 + lm) * NH;
  const bf16_t* wB2 = whb + (size_t)(i*G3 + (wave + 32 )*16 + lm) * NH;

  __syncthreads();

  #pragma unroll 1
  for (int t = 0; t < NT; ++t) {
    if (tid < NB) xcol[tid] = x[(tid * NT + t) * NI + i];

    f32x8 acc[4][3];
    #pragma unroll
    for (int mt = 0; mt < 4; ++mt)
      #pragma unroll
      for (int g = 0; g < 3; ++g)
        acc[mt][g] = (f32x8){0.f,0.f,0.f,0.f,0.f,0.f,0.f,0.f};

    // K loop: gh[64 x 768] = h[64 x 256] * wh_i^T[256 x 768]
    #pragma unroll 1
    for (int kt = 0; kt < 8; ++kt) {
      const int kbA = kt*32 + ((lane & 16) ? 8  : 0);   // A: K chunks at kbA, kbA+16
      const int kbB = kt*32 + ((lane & 16) ? 8  : 0);   // B: K chunks at kbB, kbB+16
      union { bf16x16 v; bf16x8 h[2]; } b0, b1, b2;
      b0.h[0] = *reinterpret_cast<const bf16x8*>(wB0 + kbB);
      b0.h[1] = *reinterpret_cast<const bf16x8*>(wB0 + kbB + 16);
      b1.h[0] = *reinterpret_cast<const bf16x8*>(wB1 + kbB);
      b1.h[1] = *reinterpret_cast<const bf16x8*>(wB1 + kbB + 16);
      b2.h[0] = *reinterpret_cast<const bf16x8*>(wB2 + kbB);
      b2.h[1] = *reinterpret_cast<const bf16x8*>(wB2 + kbB + 16);
      #pragma unroll
      for (int mt = 0; mt < 4; ++mt) {
        union { bf16x16 v; bf16x8 h[2]; } a;
        const bf16_t* ap = &hsh[(mt*16 + lm) * HSTR + kbA];
        a.h[0] = *reinterpret_cast<const bf16x8*>(ap);
        a.h[1] = *reinterpret_cast<const bf16x8*>(ap + 16);
        acc[mt][0] = __builtin_amdgcn_wmma_f32_16x16x32_bf16(false, a.v, false, b0.v, (short)0, acc[mt][0], false, false);
        acc[mt][1] = __builtin_amdgcn_wmma_f32_16x16x32_bf16(false, a.v, false, b1.v, (short)0, acc[mt][1], false, false);
        acc[mt][2] = __builtin_amdgcn_wmma_f32_16x16x32_bf16(false, a.v, false, b2.v, (short)0, acc[mt][2], false, false);
      }
    }

    __syncthreads();  // all GEMM reads of hsh done; xcol visible

    // register-local gate update: D layout row = mt*16 + rofs + c, col = hc
    #pragma unroll
    for (int mt = 0; mt < 4; ++mt) {
      #pragma unroll
      for (int c = 0; c < 8; ++c) {
        const int b = mt*16 + rofs + c;
        const int e = mt*8 + c;
        const float xv = xcol[b];
        const float r  = fast_sigmoid((xv*wir + bir ) + (acc[mt][0][c] + bhr));
        const float z  = fast_sigmoid((xv*wiz + biz ) + (acc[mt][1][c] + bhz));
        const float n  = fast_tanh(   (xv*win + bin_) + r * (acc[mt][2][c] + bhn));
        const float hv = (1.0f - z) * n + z * hreg[e];
        hreg[e] = hv;
        hsh[b * HSTR + hc] = (bf16_t)hv;
        if (t == lsh[b] - 1) ht[((size_t)b * NI + i) * NH + hc] = hv;  // gather at lens-1
      }
    }
    __syncthreads();  // h updated before next timestep's GEMM
  }
}

// ---------------- final attention head (MHA/FFN are dead code in the reference) ----------------
// e[i] = posi[i] . (fk @ iq) + iq.bfk ;  wctx = (sum_i a_i posi_i) @ fv + bfv (softmax sums to 1)
__global__ void __launch_bounds__(256)
head_kernel(const float* __restrict__ ht,                       // [B,I,H]
            const float* __restrict__ fq,  const float* __restrict__ bfq,
            const float* __restrict__ fk,  const float* __restrict__ bfk,
            const float* __restrict__ fv,  const float* __restrict__ bfv,
            const float* __restrict__ wout, const float* __restrict__ bout,
            float* __restrict__ out)                            // [B,2]
{
  __shared__ float mp[NH], iq[NH], u[NH], pbar[NH], wctx[NH], ea[NI];
  __shared__ float c0s;
  const int b = blockIdx.x;
  const int tid = threadIdx.x;             // 0..255 == H
  const float* pb = ht + (size_t)b * NI * NH;

  { // mean over features
    float s = 0.f;
    for (int ii = 0; ii < NI; ++ii) s += pb[ii*NH + tid];
    mp[tid] = s * (1.0f / NI);
  }
  __syncthreads();
  { // iq = mp @ fq + bfq
    float s = bfq[tid];
    for (int h = 0; h < NH; ++h) s += mp[h] * fq[h*NH + tid];
    iq[tid] = s;
  }
  __syncthreads();
  { // u = fk @ iq
    float s = 0.f;
    const float* row = fk + (size_t)tid * NH;
    for (int g = 0; g < NH; ++g) s += row[g] * iq[g];
    u[tid] = s;
  }
  if (tid == 0) {
    float s = 0.f;
    for (int g = 0; g < NH; ++g) s += iq[g] * bfk[g];
    c0s = s;
  }
  __syncthreads();
  if (tid < NI) { // scores
    float s = c0s;
    const float* prow = pb + (size_t)tid * NH;
    for (int h = 0; h < NH; ++h) s += prow[h] * u[h];
    ea[tid] = s;
  }
  __syncthreads();
  if (tid == 0) { // softmax over 64 features
    float mx = ea[0];
    for (int ii = 1; ii < NI; ++ii) mx = fmaxf(mx, ea[ii]);
    float ssum = 0.f;
    for (int ii = 0; ii < NI; ++ii) { float v = __expf(ea[ii] - mx); ea[ii] = v; ssum += v; }
    const float inv = fast_rcp(ssum);
    for (int ii = 0; ii < NI; ++ii) ea[ii] *= inv;
  }
  __syncthreads();
  { // pbar = sum_i a_i * posi_i
    float s = 0.f;
    for (int ii = 0; ii < NI; ++ii) s += ea[ii] * pb[ii*NH + tid];
    pbar[tid] = s;
  }
  __syncthreads();
  { // wctx = pbar @ fv + bfv
    float s = bfv[tid];
    for (int h = 0; h < NH; ++h) s += pbar[h] * fv[h*NH + tid];
    wctx[tid] = s;
  }
  __syncthreads();
  if (tid < 2) { // logits
    float s = bout[tid];
    for (int g = 0; g < NH; ++g) s += wctx[g] * wout[g*2 + tid];
    out[b*2 + tid] = s;
  }
}

extern "C" void kernel_launch(void* const* d_in, const int* in_sizes, int n_in,
                              void* d_out, int out_size, void* d_ws, size_t ws_size,
                              hipStream_t stream) {
  (void)in_sizes; (void)n_in; (void)out_size; (void)ws_size;

  const float* x    = (const float*)d_in[0];
  const int*   lens = (const int*)  d_in[1];
  // d_in[2] mask: all-ones, unused
  const float* wi   = (const float*)d_in[3];
  const float* bi   = (const float*)d_in[4];
  const float* wh   = (const float*)d_in[5];
  const float* bh   = (const float*)d_in[6];
  // d_in[7..18]: MHA + FFN weights -> dead code for the output
  const float* fq   = (const float*)d_in[19];
  const float* bfq  = (const float*)d_in[20];
  const float* fk   = (const float*)d_in[21];
  const float* bfk  = (const float*)d_in[22];
  const float* fv   = (const float*)d_in[23];
  const float* bfv  = (const float*)d_in[24];
  const float* wout = (const float*)d_in[25];
  const float* bout = (const float*)d_in[26];

  const int n_wh = NI * G3 * NH;                       // 12,582,912 elements
  bf16_t* whb = (bf16_t*)d_ws;                         // 25.2 MB (L2-resident)
  float*  ht  = (float*)((char*)d_ws + (size_t)n_wh * sizeof(bf16_t)); // 4 MB

  cvt_bf16_kernel<<<8192, 256, 0, stream>>>(wh, whb, n_wh);
  gru_wmma_kernel<<<NI, 512, 0, stream>>>(x, lens, wi, bi, whb, bh, ht);
  head_kernel<<<NB, 256, 0, stream>>>(ht, fq, bfq, fk, bfk, fv, bfv, wout, bout, (float*)d_out);
}
